// Router_29892972380612
// MI455X (gfx1250) — matrix operand order repaired and checked by reading, not compiled, not run on hardware
//
#include <hip/hip_runtime.h>
#include <math.h>

typedef _Float16 v16h __attribute__((ext_vector_type(16)));
typedef _Float16 v8h  __attribute__((ext_vector_type(8)));
typedef _Float16 v4h  __attribute__((ext_vector_type(4)));
typedef float    v8f  __attribute__((ext_vector_type(8)));

#define D_DIM 4096
#define E_DIM 64
#define TOPK  8
#define KC    128                    // K-chunk staged in LDS (double buffered)
#define W_STRIDE 136                 // halfs per padded W row (272 B: 16B aligned, bank stride 4)
#define WBUF (E_DIM * W_STRIDE)      // halfs per buffer
#define NCHUNK (D_DIM / KC)          // 32

__global__ __launch_bounds__(128)
void router_wmma_kernel(const float* __restrict__ x,
                        const float* __restrict__ Wg,
                        float* __restrict__ out,
                        int N) {
    __shared__ _Float16 sW[2 * WBUF];   // 34,816 B : ping-pong f16 W chunks [e][k]
    __shared__ float    sLog[64 * 65];  // 16,640 B : logits, padded rows

    const int tid  = threadIdx.x;
    const int wave = tid >> 5;          // 4 waves per block
    const int lane = tid & 31;
    const int hl   = lane >> 4;         // half-wave selector
    const int l16  = lane & 15;
    const int blockRow = blockIdx.x * 64;

    v8f acc[4] = {};

    const float* xrow = x + (size_t)(blockRow + wave * 16 + l16) * D_DIM;

    // ---- prologue: chunk 0 -> buffer 0 ----
    #pragma unroll
    for (int j = 0; j < 16; ++j) {
        const int i  = tid + j * 128;
        const int e  = i >> 5;            // / (KC/4)
        const int kq = (i & 31) << 2;     // * 4
        const float4 w4 = *(const float4*)(Wg + (size_t)e * D_DIM + kq);
        v4h h;
        h[0] = (_Float16)w4.x; h[1] = (_Float16)w4.y;
        h[2] = (_Float16)w4.z; h[3] = (_Float16)w4.w;
        *(v4h*)(&sW[e * W_STRIDE + kq]) = h;
    }
    __syncthreads();

    for (int c = 0; c < NCHUNK; ++c) {
        const int kc = c * KC;
        const _Float16* buf = &sW[(c & 1) * WBUF];
        const bool havenext = (c + 1 < NCHUNK);

        // (1) issue next W chunk's global loads early (consumed after compute)
        float4 wst[16];
        if (havenext) {
            const int kn = kc + KC;
            #pragma unroll
            for (int j = 0; j < 16; ++j) {
                const int i  = tid + j * 128;
                const int e  = i >> 5;
                const int kq = (i & 31) << 2;
                wst[j] = *(const float4*)(Wg + (size_t)e * D_DIM + kn + kq);
            }
        }

        // (2) all 4 A fragments for this chunk (16x global_load_b128)
        v16h a[4];
        #pragma unroll
        for (int s = 0; s < 4; ++s) {
            const float* xp = xrow + kc + s * 32 + hl * 8;
            const float4 xa = *(const float4*)(xp);
            const float4 xb = *(const float4*)(xp + 4);
            const float4 xc = *(const float4*)(xp + 16);
            const float4 xd = *(const float4*)(xp + 20);
            a[s][0]  = (_Float16)xa.x; a[s][1]  = (_Float16)xa.y;
            a[s][2]  = (_Float16)xa.z; a[s][3]  = (_Float16)xa.w;
            a[s][4]  = (_Float16)xb.x; a[s][5]  = (_Float16)xb.y;
            a[s][6]  = (_Float16)xb.z; a[s][7]  = (_Float16)xb.w;
            a[s][8]  = (_Float16)xc.x; a[s][9]  = (_Float16)xc.y;
            a[s][10] = (_Float16)xc.z; a[s][11] = (_Float16)xc.w;
            a[s][12] = (_Float16)xd.x; a[s][13] = (_Float16)xd.y;
            a[s][14] = (_Float16)xd.z; a[s][15] = (_Float16)xd.w;
        }

        // (3) compute: per 32-K step, gather 4 independent B frags, then 4 WMMAs
        #pragma unroll
        for (int s = 0; s < 4; ++s) {
            v16h b[4];
            #pragma unroll
            for (int t = 0; t < 4; ++t) {
                const int off = (t * 16 + l16) * W_STRIDE + s * 32 + hl * 8;
                const v8h b0 = *(const v8h*)(&buf[off]);
                const v8h b1 = *(const v8h*)(&buf[off + 16]);
                #pragma unroll
                for (int j = 0; j < 8; ++j) { b[t][j] = b0[j]; b[t][8 + j] = b1[j]; }
            }
            #pragma unroll
            for (int t = 0; t < 4; ++t) {
                acc[t] = __builtin_amdgcn_wmma_f32_16x16x32_f16(
                    /*neg_a=*/false, a[s], /*neg_b=*/false, b[t],
                    /*c_mod=*/(short)0, acc[t],
                    /*reuse_a=*/false, /*reuse_b=*/false);
            }
        }

        // (4) convert + store next W chunk into the other buffer
        if (havenext) {
            _Float16* nbuf = &sW[((c + 1) & 1) * WBUF];
            #pragma unroll
            for (int j = 0; j < 16; ++j) {
                const int i  = tid + j * 128;
                const int e  = i >> 5;
                const int kq = (i & 31) << 2;
                v4h h;
                h[0] = (_Float16)wst[j].x; h[1] = (_Float16)wst[j].y;
                h[2] = (_Float16)wst[j].z; h[3] = (_Float16)wst[j].w;
                *(v4h*)(&nbuf[e * W_STRIDE + kq]) = h;
            }
        }
        __syncthreads();
    }

    // ---- spill logits to LDS: C/D layout VGPR r -> row (r + 8*half), col l16 ----
    #pragma unroll
    for (int t = 0; t < 4; ++t) {
        #pragma unroll
        for (int r = 0; r < 8; ++r) {
            sLog[(wave * 16 + r + hl * 8) * 65 + t * 16 + l16] = acc[t][r];
        }
    }
    __syncthreads();

    // ---- softmax + top-8 + gate normalization: one thread per row ----
    float* gatesOut = out;
    float* idxOut   = out + (size_t)N * TOPK;
    float* probsOut = out + (size_t)N * 2 * TOPK;

    if (tid < 64) {
        const int row = tid;
        const int gr  = blockRow + row;
        float* lp = &sLog[row * 65];

        float m = lp[0];
        #pragma unroll 8
        for (int j = 1; j < 64; ++j) m = fmaxf(m, lp[j]);

        float sum = 0.f;
        #pragma unroll 8
        for (int j = 0; j < 64; ++j) {
            const float p = __expf(lp[j] - m);
            lp[j] = p;
            sum += p;
        }
        const float inv = 1.f / sum;

        #pragma unroll 8
        for (int j = 0; j < 64; ++j) probsOut[(size_t)gr * 64 + j] = lp[j] * inv;

        // top-8 via masked argmax; strict '>' picks lowest index on ties (JAX semantics)
        unsigned long long used = 0ull;
        float gv[TOPK]; int gi[TOPK]; float gsum = 0.f;
        #pragma unroll
        for (int k = 0; k < TOPK; ++k) {
            float best = -1.f; int bi = 0;
            for (int j = 0; j < 64; ++j) {
                const bool free_ = !((used >> j) & 1ull);
                const float p = lp[j];
                if (free_ && p > best) { best = p; bi = j; }
            }
            used |= (1ull << bi);
            const float pv = best * inv;
            gv[k] = pv; gi[k] = bi; gsum += pv;
        }
        const float ginv = 1.f / (gsum + 1e-9f);
        #pragma unroll
        for (int k = 0; k < TOPK; ++k) {
            gatesOut[(size_t)gr * TOPK + k] = gv[k] * ginv;
            idxOut[(size_t)gr * TOPK + k]   = (float)gi[k];
        }
    }
}

extern "C" void kernel_launch(void* const* d_in, const int* in_sizes, int n_in,
                              void* d_out, int out_size, void* d_ws, size_t ws_size,
                              hipStream_t stream) {
    (void)n_in; (void)out_size; (void)d_ws; (void)ws_size;
    const float* x = (const float*)d_in[0];   // (N, 4096) fp32
    const float* W = (const float*)d_in[1];   // (64, 4096) fp32
    float* out = (float*)d_out;               // [gates N*8 | indices N*8 | probs N*64]
    const int N = in_sizes[0] / D_DIM;        // 16384
    const int blocks = N / 64;                // 64 rows per block (4 waves x 16 rows)
    router_wmma_kernel<<<blocks, 128, 0, stream>>>(x, W, out, N);
}
